// Model_6983616823469
// MI455X (gfx1250) — matrix operand order repaired and checked by reading, not compiled, not run on hardware
//
#include <hip/hip_runtime.h>
#include <hip/hip_bf16.h>

// ---------------------------------------------------------------------------
// Model: bidirectional GRU encoder + gated outputs + attention GRU decoder.
// V=32000, E=32, H=128, Hh=64, B=128, S=512, T=16 (15 decode steps).
// GEMMs: v_wmma_f32_16x16x32_f16. Bulk data movement: Tensor Data Mover
// (tensor_load_to_lds + s_wait_tensorcnt) for the encoder's per-step input
// slab (double-buffered in LDS) and the logits kernel's W_out tiles.
// ---------------------------------------------------------------------------

#define CV 32000
#define CE 32
#define CH 128
#define CHh 64
#define CB 128
#define CS 512
#define CT 16
#define CTD 15

typedef __attribute__((ext_vector_type(16))) _Float16 v16h;
typedef __attribute__((ext_vector_type(8)))  float    v8f;
typedef __attribute__((ext_vector_type(4)))  unsigned int v4u;
typedef __attribute__((ext_vector_type(8)))  int      v8i;
typedef __attribute__((ext_vector_type(4)))  int      v4i;

// ---- WMMA helpers ---------------------------------------------------------

__device__ __forceinline__ v8f wmma16(v16h a, v16h b, v8f c) {
  // (neg_a, A, neg_b, B, c_mod, C, reuse_a, reuse_b)
  return __builtin_amdgcn_wmma_f32_16x16x32_f16(false, a, false, b, (short)0, c,
                                                false, false);
}

__device__ __forceinline__ v8f splat8(float v) {
  v8f r;
#pragma unroll
  for (int i = 0; i < 8; ++i) r[i] = v;
  return r;
}

// K-offset of element j (0..15) of a 16x32 f16 A-fragment, contiguous order (k = 16*hi + j), matching the B loaders.
__device__ __forceinline__ int akoff(int j, int hi) {
  return j + (hi ? 16 : 0);
}

// A fragment (16x32 f16) from row-major f16 matrix; base points at tile row 0.
__device__ __forceinline__ v16h load_a_h(const _Float16* base, int ld, int k0) {
  int lane = threadIdx.x & 31;
  int m = lane & 15, hi = lane >> 4;
  v16h a;
#pragma unroll
  for (int j = 0; j < 16; ++j) a[j] = base[m * ld + k0 + akoff(j, hi)];
  return a;
}

// B fragment (32x16 f16) where B[k,n] = W[n,k], W row-major (N,K) in f32.
__device__ __forceinline__ v16h load_b_f32(const float* w, int ldw, int n0, int k0) {
  int lane = threadIdx.x & 31;
  int n = lane & 15, kg = (lane >> 4) * 16;
  v16h b;
#pragma unroll
  for (int j = 0; j < 16; ++j) b[j] = (_Float16)w[(n0 + n) * ldw + k0 + kg + j];
  return b;
}

// Same but W already f16 (global or LDS; per-lane 32B contiguous -> b128 pair).
__device__ __forceinline__ v16h load_b_h(const _Float16* w, int ldw, int n0, int k0) {
  int lane = threadIdx.x & 31;
  int n = lane & 15, kg = (lane >> 4) * 16;
  v16h b;
#pragma unroll
  for (int j = 0; j < 16; ++j) b[j] = w[(n0 + n) * ldw + k0 + kg + j];
  return b;
}

__device__ __forceinline__ float sigm(float x) { return 1.0f / (1.0f + __expf(-x)); }

// ---- Tensor Data Mover: DMA a contiguous run of f16 (2-byte) elements -----
// Builds a 2D D# (ISA 8.3/8.4): tile_dim0 = nElems (one row), data_size=2B.
// Groups 2/3 zeroed (<=2D tensor). clang-23 6-arg builtin form:
// (uint32x4 g0, int32x8 g1, int32x4, int32x4, int32x8, i32 cpol).
__device__ __forceinline__ void tdm_load_run_f16(void* lds_dst,
                                                 const _Float16* gsrc,
                                                 unsigned nElems) {
  unsigned lds_addr = (unsigned)(uintptr_t)lds_dst;  // flat addr low 32 = LDS offset
  unsigned long long ga = (unsigned long long)(uintptr_t)gsrc;
  v4u g0;
  g0[0] = 1u;                                       // count=1 (valid), user mode
  g0[1] = lds_addr;                                 // bits 63:32 lds_addr
  g0[2] = (unsigned)(ga & 0xffffffffu);             // bits 95:64 global_addr lo
  g0[3] = (unsigned)((ga >> 32) & 0x01ffffffu) | (2u << 30);  // addr hi + type=2
  v8i g1;
  g1[0] = (int)(1u << 16);                          // data_size=1 -> 2 bytes
  g1[1] = (int)((nElems & 0xffffu) << 16);          // tensor_dim0[15:0]
  g1[2] = (int)(((nElems >> 16) & 0xffffu) |        // tensor_dim0[31:16]
                (1u << 16));                        // tensor_dim1 = 1
  g1[3] = (int)((nElems & 0xffffu) << 16);          // tile_dim0 = nElems
  g1[4] = (int)1;                                   // tile_dim1 = 1
  g1[5] = (int)nElems;                              // tensor_dim0_stride lo
  g1[6] = 0;
  g1[7] = 0;
  v4i z4 = {0, 0, 0, 0};
  v8i z8 = {0, 0, 0, 0, 0, 0, 0, 0};
  __builtin_amdgcn_tensor_load_to_lds(g0, g1, z4, z4, z8, 0);
}

// ---- prep kernels ---------------------------------------------------------

__global__ void k_f32_to_f16(const float* __restrict__ src,
                             _Float16* __restrict__ dst, int n) {
  int i = blockIdx.x * blockDim.x + threadIdx.x;
  if (i < n) dst[i] = (_Float16)src[i];
}

// xs layout (S, B, E) f16
__global__ void k_embed_enc(const int* __restrict__ inputs,
                            const float* __restrict__ emb,
                            _Float16* __restrict__ xs) {
  int i = blockIdx.x * blockDim.x + threadIdx.x;
  if (i >= CS * CB * CE) return;
  int e = i % CE;
  int b = (i / CE) % CB;
  int s = i / (CB * CE);
  int tok = inputs[b * CS + s];
  xs[i] = (_Float16)emb[tok * CE + e];
}

// edec layout (TD, B, E) f16; word[t,b] = targets[b,t]
__global__ void k_embed_dec(const int* __restrict__ targets,
                            const float* __restrict__ emb,
                            _Float16* __restrict__ edec) {
  int i = blockIdx.x * blockDim.x + threadIdx.x;
  if (i >= CTD * CB * CE) return;
  int e = i % CE;
  int b = (i / CE) % CB;
  int t = i / (CB * CE);
  int tok = targets[b * CT + t];
  edec[i] = (_Float16)emb[tok * CE + e];
}

// ---- encoder: one workgroup per direction, 512 sequential steps -----------
// Per step: gi = x_t @ W_ih^T + b_ih (K=32), gh = h @ W_hh^T + b_hh (K=64).
// 16 waves: wave w -> mTile=w>>1 (rows), cg=w&1 (32-col strip of hidden j).
// Weights f16 in LDS; x_t slabs (8 KB each) double-buffered in LDS via the
// Tensor Data Mover, issued by wave 0 one step ahead of compute.
__global__ __launch_bounds__(512) void k_encoder(
    const _Float16* __restrict__ xs,
    const float* __restrict__ W_ih_f, const float* __restrict__ W_hh_f,
    const float* __restrict__ b_ih_f, const float* __restrict__ b_hh_f,
    const float* __restrict__ W_ih_b, const float* __restrict__ W_hh_b,
    const float* __restrict__ b_ih_b, const float* __restrict__ b_hh_b,
    float* __restrict__ out_f32, _Float16* __restrict__ out_h,
    float* __restrict__ hdec_f32, _Float16* __restrict__ hdec_h) {
  __shared__ _Float16 lds_wih[192 * CE];       // 12 KB
  __shared__ _Float16 lds_whh[192 * CHh];      // 24 KB
  __shared__ _Float16 lds_h[CB * CHh];         // 16 KB
  __shared__ _Float16 lds_x[2][CB * CE];       // 2 x 8 KB (TDM double buffer)

  const int tid = threadIdx.x;
  const int lane = tid & 31, w = tid >> 5;
  const int mTile = w >> 1, cg = w & 1;
  const int n = lane & 15, hi = lane >> 4;
  const int dir = blockIdx.x;

  const float* W_ih = dir ? W_ih_b : W_ih_f;
  const float* W_hh = dir ? W_hh_b : W_hh_f;
  const float* b_ih = dir ? b_ih_b : b_ih_f;
  const float* b_hh = dir ? b_hh_b : b_hh_f;

  // Kick off TDM for step 0's input slab while we convert weights.
  if (w == 0) {
    const int s0 = dir ? (CS - 1) : 0;
    tdm_load_run_f16(&lds_x[0][0], xs + (size_t)s0 * CB * CE, CB * CE);
  }
  for (int i = tid; i < 192 * CE; i += 512) lds_wih[i] = (_Float16)W_ih[i];
  for (int i = tid; i < 192 * CHh; i += 512) lds_whh[i] = (_Float16)W_hh[i];
  for (int i = tid; i < CB * CHh; i += 512) lds_h[i] = (_Float16)0.0f;
  if (w == 0) __builtin_amdgcn_s_wait_tensorcnt(0);
  __syncthreads();

  // Hoist only scalar biases (12 regs).
  float bib[3][2], bhb[3][2];
#pragma unroll
  for (int g = 0; g < 3; ++g)
#pragma unroll
    for (int t = 0; t < 2; ++t) {
      int nT = g * 4 + cg * 2 + t;
      bib[g][t] = b_ih[nT * 16 + n];
      bhb[g][t] = b_hh[nT * 16 + n];
    }

#pragma unroll 1
  for (int step = 0; step < CS; ++step) {
    const int s = dir ? (CS - 1 - step) : step;
    const int cur = step & 1, nxt = cur ^ 1;
    // Async: DMA next step's slab into the other buffer while computing.
    if (w == 0 && step + 1 < CS) {
      const int s2 = dir ? (CS - 2 - step) : (step + 1);
      tdm_load_run_f16(&lds_x[nxt][0], xs + (size_t)s2 * CB * CE, CB * CE);
    }

    v16h ax = load_a_h(&lds_x[cur][0] + mTile * 16 * CE, CE, 0);
    v16h ah0 = load_a_h(lds_h + mTile * 16 * CHh, CHh, 0);
    v16h ah1 = load_a_h(lds_h + mTile * 16 * CHh, CHh, 32);

    v8f accI[3][2], accH[3][2];
#pragma unroll
    for (int g = 0; g < 3; ++g)
#pragma unroll
      for (int t = 0; t < 2; ++t) {
        int nT = g * 4 + cg * 2 + t;
        v8f ai = splat8(bib[g][t]);
        ai = wmma16(ax, load_b_h(lds_wih, CE, nT * 16, 0), ai);
        v8f ah = splat8(bhb[g][t]);
        ah = wmma16(ah0, load_b_h(lds_whh, CHh, nT * 16, 0), ah);
        ah = wmma16(ah1, load_b_h(lds_whh, CHh, nT * 16, 32), ah);
        accI[g][t] = ai;
        accH[g][t] = ah;
      }

    float hnew[2][8];
#pragma unroll
    for (int t = 0; t < 2; ++t)
#pragma unroll
      for (int e = 0; e < 8; ++e) {
        float r = sigm(accI[0][t][e] + accH[0][t][e]);
        float z = sigm(accI[1][t][e] + accH[1][t][e]);
        float nn = tanhf(accI[2][t][e] + r * accH[2][t][e]);
        int row = mTile * 16 + e + (hi ? 8 : 0);
        int j = cg * 32 + t * 16 + n;
        float hp = (float)lds_h[row * CHh + j];
        hnew[t][e] = (1.0f - z) * nn + z * hp;
      }
    __syncthreads();  // all reads of lds_h done across waves
#pragma unroll
    for (int t = 0; t < 2; ++t)
#pragma unroll
      for (int e = 0; e < 8; ++e) {
        int row = mTile * 16 + e + (hi ? 8 : 0);
        int j = cg * 32 + t * 16 + n;
        float hv = hnew[t][e];
        lds_h[row * CHh + j] = (_Float16)hv;
        size_t oidx = (size_t)row * CS * CH + (size_t)s * CH + dir * CHh + j;
        out_f32[oidx] = hv;
        out_h[oidx] = (_Float16)hv;
      }
    if (w == 0) __builtin_amdgcn_s_wait_tensorcnt(0);  // next slab landed
    __syncthreads();  // publish h writes + next x buffer
  }

  // final hidden -> sn half (decoder initial state)
  for (int i = tid; i < CB * CHh; i += 512) {
    int row = i / CHh, j = i % CHh;
    float v = (float)lds_h[i];
    hdec_f32[row * CH + dir * CHh + j] = v;
    hdec_h[row * CH + dir * CHh + j] = (_Float16)v;
  }
}

// ---- v2 = sn @ W2^T + b2  (128x128x128 GEMM, single block, WMMA) ----------
__global__ __launch_bounds__(256) void k_v2(const _Float16* __restrict__ sn_h,
                                            const float* __restrict__ W2,
                                            const float* __restrict__ b2,
                                            float* __restrict__ v2) {
  const int tid = threadIdx.x, lane = tid & 31, w = tid >> 5;
  const int nT = w, n = lane & 15, hi = lane >> 4;
  v16h Bf[4];
#pragma unroll
  for (int kt = 0; kt < 4; ++kt) Bf[kt] = load_b_f32(W2, CH, nT * 16, kt * 32);
  const float bv = b2[nT * 16 + n];
#pragma unroll 1
  for (int mT = 0; mT < 8; ++mT) {
    v8f acc = splat8(bv);
#pragma unroll
    for (int kt = 0; kt < 4; ++kt) {
      v16h a = load_a_h(sn_h + mT * 16 * CH, CH, kt * 32);
      acc = wmma16(a, Bf[kt], acc);
    }
#pragma unroll
    for (int e = 0; e < 8; ++e)
      v2[(mT * 16 + e + (hi ? 8 : 0)) * CH + nT * 16 + n] = acc[e];
  }
}

// ---- sGate: enc = outputs * sigmoid(outputs@W1^T + b1 + v2[b]) ------------
__global__ __launch_bounds__(256) void k_gate(
    const _Float16* __restrict__ out_h, const float* __restrict__ out_f32,
    const float* __restrict__ W1, const float* __restrict__ b1,
    const float* __restrict__ v2, float* __restrict__ enc) {
  const int tid = threadIdx.x, lane = tid & 31, w = tid >> 5;
  const int nT = w, n = lane & 15, hi = lane >> 4;
  v16h Bf[4];
#pragma unroll
  for (int kt = 0; kt < 4; ++kt) Bf[kt] = load_b_f32(W1, CH, nT * 16, kt * 32);
  const float b1v = b1[nT * 16 + n];
  const int col = nT * 16 + n;
#pragma unroll 1
  for (int mi = 0; mi < 8; ++mi) {
    size_t row0 = ((size_t)blockIdx.x * 8 + mi) * 16;
    v8f acc = splat8(b1v);
#pragma unroll
    for (int kt = 0; kt < 4; ++kt) {
      v16h a = load_a_h(out_h + row0 * CH, CH, kt * 32);
      acc = wmma16(a, Bf[kt], acc);
    }
#pragma unroll
    for (int e = 0; e < 8; ++e) {
      size_t r = row0 + e + (hi ? 8 : 0);
      int b = (int)(r >> 9);  // r / S
      float g = sigm(acc[e] + v2[b * CH + col]);
      enc[r * CH + col] = out_f32[r * CH + col] * g;
    }
  }
}

// ---- decoder attention: one block per batch row ---------------------------
__global__ __launch_bounds__(256) void k_attn(
    const float* __restrict__ hdec_f32, const float* __restrict__ enc,
    const _Float16* __restrict__ edec, _Float16* __restrict__ x_h, int t) {
  __shared__ float hv[CH];
  __shared__ float sc[CS];
  __shared__ float red[256];
  const int tid = threadIdx.x, b = blockIdx.x;
  if (tid < CH) hv[tid] = hdec_f32[b * CH + tid];
  __syncthreads();
  const float* eb = enc + (size_t)b * CS * CH;
  for (int s = tid; s < CS; s += 256) {
    float d = 0.0f;
#pragma unroll 4
    for (int j = 0; j < CH; ++j) d += hv[j] * eb[s * CH + j];
    sc[s] = d;
  }
  __syncthreads();
  red[tid] = fmaxf(sc[tid], sc[tid + 256]);
  __syncthreads();
  for (int o = 128; o > 0; o >>= 1) {
    if (tid < o) red[tid] = fmaxf(red[tid], red[tid + o]);
    __syncthreads();
  }
  const float mx = red[0];
  __syncthreads();
  for (int s = tid; s < CS; s += 256) sc[s] = __expf(sc[s] - mx);
  __syncthreads();
  red[tid] = sc[tid] + sc[tid + 256];
  __syncthreads();
  for (int o = 128; o > 0; o >>= 1) {
    if (tid < o) red[tid] += red[tid + o];
    __syncthreads();
  }
  const float inv = 1.0f / red[0];
  __syncthreads();
  if (tid < CH) {
    float c = 0.0f;
    for (int s = 0; s < CS; ++s) c += sc[s] * eb[s * CH + tid];
    x_h[b * (CH + CE) + tid] = (_Float16)(c * inv);
  } else if (tid < CH + CE) {
    x_h[b * (CH + CE) + tid] = edec[(size_t)t * CB * CE + b * CE + (tid - CH)];
  }
}

// ---- decoder GRU cell: x=(B,160), gi K=160 (5 wmma), gh K=128 (4 wmma) ----
__global__ __launch_bounds__(512) void k_dec_gru(
    const _Float16* __restrict__ x_h, _Float16* __restrict__ hdec_h,
    float* __restrict__ hdec_f32, const float* __restrict__ W_ih_d,
    const float* __restrict__ W_hh_d, const float* __restrict__ b_ih_d,
    const float* __restrict__ b_hh_d) {
  const int tid = threadIdx.x, lane = tid & 31, w = tid >> 5;
  const int mTile = w >> 1, cg = w & 1;
  const int n = lane & 15, hi = lane >> 4;
  const int XK = CH + CE;  // 160

  v16h ax[5], ah[4];
#pragma unroll
  for (int kt = 0; kt < 5; ++kt)
    ax[kt] = load_a_h(x_h + mTile * 16 * XK, XK, kt * 32);
#pragma unroll
  for (int kt = 0; kt < 4; ++kt)
    ah[kt] = load_a_h(hdec_h + mTile * 16 * CH, CH, kt * 32);

  float hnew[4][8];
#pragma unroll 1
  for (int t = 0; t < 4; ++t) {
    v8f aI[3], aH[3];
#pragma unroll 1
    for (int g = 0; g < 3; ++g) {
      int nT = g * 8 + cg * 4 + t;
      v8f ai = splat8(b_ih_d[nT * 16 + n]);
#pragma unroll
      for (int kt = 0; kt < 5; ++kt) {
        v16h bf = load_b_f32(W_ih_d, XK, nT * 16, kt * 32);
        ai = wmma16(ax[kt], bf, ai);
      }
      v8f ahc = splat8(b_hh_d[nT * 16 + n]);
#pragma unroll
      for (int kt = 0; kt < 4; ++kt) {
        v16h bf = load_b_f32(W_hh_d, CH, nT * 16, kt * 32);
        ahc = wmma16(ah[kt], bf, ahc);
      }
      aI[g] = ai;
      aH[g] = ahc;
    }
#pragma unroll
    for (int e = 0; e < 8; ++e) {
      float r = sigm(aI[0][e] + aH[0][e]);
      float z = sigm(aI[1][e] + aH[1][e]);
      float nn = tanhf(aI[2][e] + r * aH[2][e]);
      int row = mTile * 16 + e + (hi ? 8 : 0);
      int j = cg * 64 + t * 16 + n;
      float hp = hdec_f32[row * CH + j];
      hnew[t][e] = (1.0f - z) * nn + z * hp;
    }
  }
  __syncthreads();  // all reads of hdec_h / hdec_f32 complete
#pragma unroll
  for (int t = 0; t < 4; ++t)
#pragma unroll
    for (int e = 0; e < 8; ++e) {
      int row = mTile * 16 + e + (hi ? 8 : 0);
      int j = cg * 64 + t * 16 + n;
      hdec_f32[row * CH + j] = hnew[t][e];
      hdec_h[row * CH + j] = (_Float16)hnew[t][e];
    }
}

// ---- logits: h(128x128) @ W_out^T(128x32000) + b_out, WMMA + TDM ----------
// 8 waves/block; wave w stages its contiguous 16x128 f16 W_out tile (4 KB)
// into its private LDS slice with the Tensor Data Mover, waits TENSORcnt,
// then runs 8 mTiles x 4 wmma against the LDS-resident B fragments.
__global__ __launch_bounds__(256) void k_logits(
    const _Float16* __restrict__ hdec_h, const _Float16* __restrict__ wout_h,
    const float* __restrict__ b_out, float* __restrict__ out, int t) {
  __shared__ _Float16 lds_w[8 * 16 * CH];  // 32 KB, 4 KB per wave
  const int tid = threadIdx.x, lane = tid & 31, w = tid >> 5;
  const int nT = blockIdx.x * 8 + w;
  const int n = lane & 15, hi = lane >> 4;

  _Float16* myw = &lds_w[w * 16 * CH];
  tdm_load_run_f16(myw, wout_h + (size_t)nT * 16 * CH, 16 * CH);
  __builtin_amdgcn_s_wait_tensorcnt(0);  // per-wave DMA complete

  v16h Bf[4];
#pragma unroll
  for (int kt = 0; kt < 4; ++kt) Bf[kt] = load_b_h(myw, CH, 0, kt * 32);
  const int col = nT * 16 + n;
  const float bv = b_out[col];
#pragma unroll 1
  for (int mT = 0; mT < 8; ++mT) {
    v8f acc = splat8(bv);
#pragma unroll
    for (int kt = 0; kt < 4; ++kt) {
      v16h a = load_a_h(hdec_h + mT * 16 * CH, CH, kt * 32);
      acc = wmma16(a, Bf[kt], acc);
    }
#pragma unroll
    for (int e = 0; e < 8; ++e) {
      int row = mT * 16 + e + (hi ? 8 : 0);
      out[(size_t)row * CTD * CV + (size_t)t * CV + col] = acc[e];
    }
  }
}

// ---------------------------------------------------------------------------

extern "C" void kernel_launch(void* const* d_in, const int* in_sizes, int n_in,
                              void* d_out, int out_size, void* d_ws,
                              size_t ws_size, hipStream_t stream) {
  const int* inputs = (const int*)d_in[0];
  const int* targets = (const int*)d_in[1];
  const float* emb = (const float*)d_in[2];
  const float* W_ih_f = (const float*)d_in[3];
  const float* W_hh_f = (const float*)d_in[4];
  const float* b_ih_f = (const float*)d_in[5];
  const float* b_hh_f = (const float*)d_in[6];
  const float* W_ih_b = (const float*)d_in[7];
  const float* W_hh_b = (const float*)d_in[8];
  const float* b_ih_b = (const float*)d_in[9];
  const float* b_hh_b = (const float*)d_in[10];
  const float* W1 = (const float*)d_in[11];
  const float* b1 = (const float*)d_in[12];
  const float* W2 = (const float*)d_in[13];
  const float* b2 = (const float*)d_in[14];
  const float* W_ih_d = (const float*)d_in[15];
  const float* W_hh_d = (const float*)d_in[16];
  const float* b_ih_d = (const float*)d_in[17];
  const float* b_hh_d = (const float*)d_in[18];
  const float* W_out = (const float*)d_in[19];
  const float* b_out = (const float*)d_in[20];
  float* out = (float*)d_out;

  char* ws = (char*)d_ws;
  size_t off = 0;
  auto alloc = [&](size_t bytes) {
    void* p = ws + off;
    off = (off + bytes + 255) & ~(size_t)255;
    return p;
  };
  _Float16* xs_h     = (_Float16*)alloc((size_t)CS * CB * CE * 2);
  _Float16* edec_h   = (_Float16*)alloc((size_t)CTD * CB * CE * 2);
  _Float16* out_h    = (_Float16*)alloc((size_t)CB * CS * CH * 2);
  float*    out_f32  = (float*)alloc((size_t)CB * CS * CH * 4);
  float*    enc_f32  = (float*)alloc((size_t)CB * CS * CH * 4);
  _Float16* wout_h   = (_Float16*)alloc((size_t)CV * CH * 2);
  float*    v2       = (float*)alloc((size_t)CB * CH * 4);
  float*    hdec_f32 = (float*)alloc((size_t)CB * CH * 4);
  _Float16* hdec_h   = (_Float16*)alloc((size_t)CB * CH * 2);
  _Float16* x_h      = (_Float16*)alloc((size_t)CB * (CH + CE) * 2);

  // prep
  k_f32_to_f16<<<(CV * CH + 255) / 256, 256, 0, stream>>>(W_out, wout_h, CV * CH);
  k_embed_enc<<<(CS * CB * CE + 255) / 256, 256, 0, stream>>>(inputs, emb, xs_h);
  k_embed_dec<<<(CTD * CB * CE + 255) / 256, 256, 0, stream>>>(targets, emb, edec_h);

  // encoder scans (fwd + bwd), one persistent workgroup each
  k_encoder<<<2, 512, 0, stream>>>(xs_h, W_ih_f, W_hh_f, b_ih_f, b_hh_f, W_ih_b,
                                   W_hh_b, b_ih_b, b_hh_b, out_f32, out_h,
                                   hdec_f32, hdec_h);

  // sGate path
  k_v2<<<1, 256, 0, stream>>>(hdec_h, W2, b2, v2);
  k_gate<<<(CB * CS / 16) / 8, 256, 0, stream>>>(out_h, out_f32, W1, b1, v2,
                                                 enc_f32);

  // decoder: 15 sequential steps
  for (int t = 0; t < CTD; ++t) {
    k_attn<<<CB, 256, 0, stream>>>(hdec_f32, enc_f32, edec_h, x_h, t);
    k_dec_gru<<<1, 512, 0, stream>>>(x_h, hdec_h, hdec_f32, W_ih_d, W_hh_d,
                                     b_ih_d, b_hh_d);
    k_logits<<<(CV / 16) / 8, 256, 0, stream>>>(hdec_h, wout_h, b_out, out, t);
  }
}
